// STConvBlock_72902774882516
// MI455X (gfx1250) — matrix operand order repaired and checked by reading, not compiled, not run on hardware
//
#include <hip/hip_runtime.h>
#include <hip/hip_bf16.h>
#include <math.h>

typedef __attribute__((ext_vector_type(16))) _Float16 v16h;
typedef __attribute__((ext_vector_type(8)))  _Float16 v8h;
typedef __attribute__((ext_vector_type(8)))  float    v8f;
typedef __attribute__((ext_vector_type(4)))  int      v4i;

#define KTT 3
#define KSS 3
#define BB  16
#define T0  32
#define T1  30
#define T2  28
#define NN  1024
#define CSI 64
#define CT  64
#define COO 128
#define KW  (KTT * CT)      /* 192: K extent of all three GEMM families */
#define LN_EPS 1e-3f

#define GLOBAL_AS __attribute__((address_space(1)))
#define LDS_AS    __attribute__((address_space(3)))

// CDNA5 async global->LDS staging (ASYNCcnt path), guarded so absence of the
// builtins falls back to direct global loads.
#if defined(__has_builtin)
#  if __has_builtin(__builtin_amdgcn_global_load_async_to_lds_b128) && \
      __has_builtin(__builtin_amdgcn_s_wait_asynccnt)
#    define USE_ASYNC_LDS 1
#  endif
#endif
#ifndef USE_ASYNC_LDS
#  define USE_ASYNC_LDS 0
#endif

static __device__ __forceinline__ v8f wmma16(v16h a, v16h b, v8f c) {
  return __builtin_amdgcn_wmma_f32_16x16x32_f16(false, a, false, b, (short)0, c, false, false);
}
static __device__ __forceinline__ float sigmoidf_(float x) {
  return 1.0f / (1.0f + __expf(-x));
}

// A tile (16x32 f16) from row-major A[m][k] (K-contiguous), lda in elements.
static __device__ __forceinline__ v16h load_a16(const _Float16* __restrict__ A, int lda,
                                                int m0, int k0, int lane) {
  const _Float16* p = A + (size_t)(m0 + (lane & 15)) * lda + k0 + ((lane >> 4) << 3);
  v8h lo = *(const v8h*)p;          // K chunk +0
  v8h hi = *(const v8h*)(p + 16);   // K chunk +16
  return __builtin_shufflevector(lo, hi, 0,1,2,3,4,5,6,7,8,9,10,11,12,13,14,15);
}

// B tile (32x16 f16) from B^T stored row-major BT[n][k] (K-contiguous), ldk in elements.
static __device__ __forceinline__ v16h load_b16T(const _Float16* __restrict__ BT, int ldk,
                                                 int n0, int k0, int lane) {
  const _Float16* p = BT + (size_t)(n0 + (lane & 15)) * ldk + k0 + ((lane >> 4) << 3);
  v8h lo = *(const v8h*)p;          // K chunk +0
  v8h hi = *(const v8h*)(p + 16);   // K chunk +16
  return __builtin_shufflevector(lo, hi, 0,1,2,3,4,5,6,7,8,9,10,11,12,13,14,15);
}

// ---------------- prep: weights f32->f16 transposed to [n][k], gk transpose ----------------
__global__ void k_prep_weights(const float* __restrict__ Wt1, const float* __restrict__ Wt3,
                               const float* __restrict__ Ws,
                               _Float16* __restrict__ Wt1hT, _Float16* __restrict__ Wt3hT,
                               _Float16* __restrict__ WshT) {
  int i = blockIdx.x * 256 + threadIdx.x;
  if (i < KW * 2 * CT)  { int k = i >> 7, n = i & 127; Wt1hT[n * KW + k] = (_Float16)Wt1[i]; }
  if (i < KW * 2 * COO) { int k = i >> 8, n = i & 255; Wt3hT[n * KW + k] = (_Float16)Wt3[i]; }
  if (i < KW * CT)      { int k = i >> 6, n = i & 63;  WshT[n * KW + k]  = (_Float16)Ws[i];  }
}

__global__ void k_transpose_gk(const float* __restrict__ gk, _Float16* __restrict__ gkT) {
  int i = blockIdx.x * 256 + threadIdx.x;        // < 1024 * 768
  int k = i / 768;                               // node (row of gk)
  int m = (i - k * 768) * 4;                     // 4 consecutive gk columns
  float4 v = *(const float4*)(gk + (size_t)k * (KSS * NN) + m);
  gkT[(size_t)(m + 0) * NN + k] = (_Float16)v.x;
  gkT[(size_t)(m + 1) * NN + k] = (_Float16)v.y;
  gkT[(size_t)(m + 2) * NN + k] = (_Float16)v.z;
  gkT[(size_t)(m + 3) * NN + k] = (_Float16)v.w;
}

// ---------------- stage A: temporal GLU #1 -> x1^T (f16, [bt][c][node]) ----------------
__global__ void __launch_bounds__(256) k_tglu1(const float* __restrict__ x,
                                               const _Float16* __restrict__ Wt1hT,
                                               const float* __restrict__ bt1,
                                               _Float16* __restrict__ x1T) {
  const int lane = threadIdx.x & 31;
  const int w    = threadIdx.x >> 5;
  const int gw   = blockIdx.x * 8 + w;
  const int ntile = gw & 63;
  const int bt    = gw >> 6;                      // 0..479
  if (bt >= BB * T1) return;
  const int b = bt / T1, t = bt % T1;
  const int m0 = ntile * 16;

  v8f accP[4] = {{}, {}, {}, {}};
  v8f accQ[4] = {{}, {}, {}, {}};
  const int mA   = m0 + (lane & 15);
  const int kloA = (lane >> 4) << 3;

  for (int k0 = 0; k0 < KW; k0 += 32) {
    v16h a;
    {   // A: x window, f32 -> f16, chunks never cross a kt block (all offsets mult of 8)
      v8h h2[2];
#pragma unroll
      for (int h = 0; h < 2; ++h) {
        int kk = k0 + kloA + h * 16;
        int kt = kk >> 6, cc = kk & 63;
        v8f f = *(const v8f*)(x + (((size_t)(b * T0 + t + kt) * NN + mA) * CSI + cc));
#pragma unroll
        for (int i = 0; i < 8; ++i) h2[h][i] = (_Float16)f[i];
      }
      a = __builtin_shufflevector(h2[0], h2[1], 0,1,2,3,4,5,6,7,8,9,10,11,12,13,14,15);
    }
#pragma unroll
    for (int pq = 0; pq < 4; ++pq) {
      v16h bp = load_b16T(Wt1hT, KW, pq * 16,      k0, lane);
      v16h bq = load_b16T(Wt1hT, KW, CT + pq * 16, k0, lane);
      accP[pq] = wmma16(a, bp, accP[pq]);
      accQ[pq] = wmma16(a, bq, accQ[pq]);
    }
  }
#pragma unroll
  for (int pq = 0; pq < 4; ++pq) {
    const int o = pq * 16 + (lane & 15);          // o in [0,64)
#pragma unroll
    for (int v = 0; v < 8; ++v) {
      int m = m0 + v + ((lane >> 4) << 3);
      float xin = x[((size_t)(b * T0 + t + 2) * NN + m) * CSI + o];
      float pv  = accP[pq][v] + bt1[o] + xin;
      float qv  = sigmoidf_(accQ[pq][v] + bt1[o + CT]);
      x1T[((size_t)bt * CT + o) * NN + m] = (_Float16)(pv * qv);
    }
  }
}

// ---------------- stage B: graph conv -> x2 (f16, [bt][node][c]) ----------------
__global__ void __launch_bounds__(256) k_graph(const _Float16* __restrict__ gkT,
                                               const _Float16* __restrict__ x1T,
                                               const _Float16* __restrict__ WshT,
                                               const float* __restrict__ bs,
                                               _Float16* __restrict__ x2h) {
  __shared__ _Float16 sA[128 * KW];               // 128 nodes x 192, 48 KB
#if USE_ASYNC_LDS
  __shared__ _Float16 sB[2][64 * 32];             // double-buffered B K-slab, 2 x 4 KB
#endif
  const int lane = threadIdx.x & 31;
  const int w    = threadIdx.x >> 5;
  const int bt   = blockIdx.x >> 3;               // 0..479
  const int m0   = (blockIdx.x & 7) * 128;        // node tile of 128
  const _Float16* x1bt = x1T + (size_t)bt * CT * NN;

  // phase 1: Gpart(3*128 x 64) = gkT[{kk*1024 + m}] @ x1; 12 WMMA per K-step.
  {
    v8f acc[3][4] = {};
    const int mbase = m0 + w * 16;
#if USE_ASYNC_LDS
    // each thread async-copies one 16B chunk of the 64x32-half slab per K-step
    const int tid    = threadIdx.x;
    const int cn     = tid >> 2;                  // slab row (c index), 0..63
    const int cchunk = (tid & 3) << 3;            // half offset within row: 0,8,16,24
    __builtin_amdgcn_global_load_async_to_lds_b128(
        (GLOBAL_AS v4i*)(x1bt + (size_t)cn * NN + cchunk),
        (LDS_AS v4i*)(&sB[0][cn * 32 + cchunk]), 0, 0);
    __builtin_amdgcn_s_wait_asynccnt(0);
    __syncthreads();
#endif
    for (int k0 = 0, it = 0; k0 < NN; k0 += 32, ++it) {
      v16h bmat[4];
#if USE_ASYNC_LDS
      if (k0 + 32 < NN) {   // prefetch next slab into the other buffer
        __builtin_amdgcn_global_load_async_to_lds_b128(
            (GLOBAL_AS v4i*)(x1bt + (size_t)cn * NN + k0 + 32 + cchunk),
            (LDS_AS v4i*)(&sB[(it + 1) & 1][cn * 32 + cchunk]), 0, 0);
      }
      const _Float16* slab = &sB[it & 1][0];
#pragma unroll
      for (int ct = 0; ct < 4; ++ct)
      {
        const _Float16* q = slab + (ct * 16 + (lane & 15)) * 32 + ((lane >> 4) << 3);
        bmat[ct] = __builtin_shufflevector(*(const v8h*)q, *(const v8h*)(q + 16), 0,1,2,3,4,5,6,7,8,9,10,11,12,13,14,15);
      }
#else
#pragma unroll
      for (int ct = 0; ct < 4; ++ct)
        bmat[ct] = load_b16T(x1bt, NN, ct * 16, k0, lane);
      __builtin_prefetch((const void*)(x1bt + (size_t)(lane & 15) * NN + k0 + 128), 0, 0);
#endif
#pragma unroll
      for (int kk = 0; kk < 3; ++kk) {
        v16h a = load_a16(gkT, NN, kk * NN + mbase, k0, lane);
#pragma unroll
        for (int ct = 0; ct < 4; ++ct)
          acc[kk][ct] = wmma16(a, bmat[ct], acc[kk][ct]);
      }
#if USE_ASYNC_LDS
      __builtin_amdgcn_s_wait_asynccnt(0);
      __syncthreads();
#endif
    }
    // scatter into LDS as A2[m_local][cc*3+kk]
#pragma unroll
    for (int kk = 0; kk < 3; ++kk)
#pragma unroll
      for (int ct = 0; ct < 4; ++ct)
#pragma unroll
        for (int v = 0; v < 8; ++v) {
          int m_local = w * 16 + v + ((lane >> 4) << 3);
          int cc = ct * 16 + (lane & 15);
          sA[m_local * KW + cc * KSS + kk] = (_Float16)acc[kk][ct][v];
        }
  }
  __syncthreads();

  // phase 2: x2 = relu(A2(128x192) @ Ws(192x64) + bs + x1); wave w owns rows w*16..w*16+15
  {
    v8f acc[4] = {{}, {}, {}, {}};
    for (int k0 = 0; k0 < KW; k0 += 32) {
      v16h a = load_a16(sA, KW, w * 16, k0, lane);
#pragma unroll
      for (int ct = 0; ct < 4; ++ct) {
        v16h bm = load_b16T(WshT, KW, ct * 16, k0, lane);
        acc[ct] = wmma16(a, bm, acc[ct]);
      }
    }
#pragma unroll
    for (int ct = 0; ct < 4; ++ct) {
      const int o = ct * 16 + (lane & 15);
#pragma unroll
      for (int v = 0; v < 8; ++v) {
        int m = m0 + w * 16 + v + ((lane >> 4) << 3);
        float val = acc[ct][v] + bs[o] + (float)x1bt[(size_t)o * NN + m];
        val = val > 0.0f ? val : 0.0f;
        x2h[((size_t)bt * NN + m) * CT + o] = (_Float16)val;
      }
    }
  }
}

// ---------------- stage C: temporal GLU #2 -> x3 (f32, [bt][node][128]) ----------------
__global__ void __launch_bounds__(256) k_tglu2(const _Float16* __restrict__ x2h,
                                               const _Float16* __restrict__ Wt3hT,
                                               const float* __restrict__ bt3,
                                               float* __restrict__ x3) {
  const int lane = threadIdx.x & 31;
  const int w    = threadIdx.x >> 5;
  const int gw   = blockIdx.x * 8 + w;
  const int ph    = gw & 1;                       // output half: o base 0 or 64
  const int ntile = (gw >> 1) & 63;
  const int bt    = gw >> 7;                      // 0..447
  if (bt >= BB * T2) return;
  const int b = bt / T2, t = bt % T2;
  const int m0 = ntile * 16;
  const int obase = ph * 64;

  v8f accP[4] = {{}, {}, {}, {}};
  v8f accQ[4] = {{}, {}, {}, {}};
  const int mA   = m0 + (lane & 15);
  const int kloA = (lane >> 4) << 3;

  for (int k0 = 0; k0 < KW; k0 += 32) {
    v16h a;
    {
      v8h h2[2];
#pragma unroll
      for (int h = 0; h < 2; ++h) {
        int kk = k0 + kloA + h * 16;
        int kt = kk >> 6, cc = kk & 63;
        h2[h] = *(const v8h*)(x2h + (((size_t)(b * T1 + t + kt) * NN + mA) * CT + cc));
      }
      a = __builtin_shufflevector(h2[0], h2[1], 0,1,2,3,4,5,6,7,8,9,10,11,12,13,14,15);
    }
#pragma unroll
    for (int pq = 0; pq < 4; ++pq) {
      v16h bp = load_b16T(Wt3hT, KW, obase + pq * 16,       k0, lane);
      v16h bq = load_b16T(Wt3hT, KW, COO + obase + pq * 16, k0, lane);
      accP[pq] = wmma16(a, bp, accP[pq]);
      accQ[pq] = wmma16(a, bq, accQ[pq]);
    }
  }
#pragma unroll
  for (int pq = 0; pq < 4; ++pq) {
    const int o = obase + pq * 16 + (lane & 15);  // o in [0,128)
#pragma unroll
    for (int v = 0; v < 8; ++v) {
      int m = m0 + v + ((lane >> 4) << 3);
      float xin = (o < CT) ? (float)x2h[((size_t)(b * T1 + t + 2) * NN + m) * CT + o] : 0.0f;
      float pv = accP[pq][v] + bt3[o] + xin;
      float qv = sigmoidf_(accQ[pq][v] + bt3[o + COO]);
      x3[((size_t)bt * NN + m) * COO + o] = pv * qv;
    }
  }
}

// ---------------- LayerNorm over (node, channel) per (b,t) ----------------
__global__ void k_ln_stats(const float* __restrict__ x3, float* __restrict__ stats) {
  __shared__ float s1[256], s2[256];
  const int bt = blockIdx.x;
  const float* p = x3 + (size_t)bt * NN * COO;
  float sum = 0.f, sq = 0.f;
  for (int i = threadIdx.x * 4; i < NN * COO; i += 256 * 4) {
    float4 v = *(const float4*)(p + i);
    sum += v.x + v.y + v.z + v.w;
    sq  += v.x * v.x + v.y * v.y + v.z * v.z + v.w * v.w;
  }
  s1[threadIdx.x] = sum; s2[threadIdx.x] = sq;
  __syncthreads();
  for (int s = 128; s > 0; s >>= 1) {
    if (threadIdx.x < s) { s1[threadIdx.x] += s1[threadIdx.x + s]; s2[threadIdx.x] += s2[threadIdx.x + s]; }
    __syncthreads();
  }
  if (threadIdx.x == 0) {
    float mean = s1[0] * (1.0f / (NN * COO));
    float var  = s2[0] * (1.0f / (NN * COO)) - mean * mean;
    stats[bt * 2]     = mean;
    stats[bt * 2 + 1] = rsqrtf(var + LN_EPS);
  }
}

__global__ void k_ln_apply(const float* __restrict__ x3, const float* __restrict__ stats,
                           const float* __restrict__ gam, const float* __restrict__ bet,
                           float* __restrict__ out) {
  size_t idx = ((size_t)blockIdx.x * 256 + threadIdx.x) * 4;
  const size_t total = (size_t)BB * T2 * NN * COO;
  if (idx >= total) return;
  int bt = (int)(idx / (NN * COO));
  int r  = (int)(idx % (NN * COO));               // node*128 + o == gamma/beta flat index
  float mean = stats[bt * 2], inv = stats[bt * 2 + 1];
  float4 v = *(const float4*)(x3 + idx);
  float4 g = *(const float4*)(gam + r);
  float4 be = *(const float4*)(bet + r);
  float4 o4;
  o4.x = g.x * (v.x - mean) * inv + be.x;
  o4.y = g.y * (v.y - mean) * inv + be.y;
  o4.z = g.z * (v.z - mean) * inv + be.z;
  o4.w = g.w * (v.w - mean) * inv + be.w;
  *(float4*)(out + idx) = o4;
}

// ---------------- driver ----------------
extern "C" void kernel_launch(void* const* d_in, const int* in_sizes, int n_in,
                              void* d_out, int out_size, void* d_ws, size_t ws_size,
                              hipStream_t stream) {
  (void)in_sizes; (void)n_in; (void)out_size; (void)ws_size;
  const float* x    = (const float*)d_in[0];
  const float* gk   = (const float*)d_in[1];
  const float* Wt1  = (const float*)d_in[2];
  const float* bt1  = (const float*)d_in[3];
  const float* Ws   = (const float*)d_in[4];
  const float* bs   = (const float*)d_in[5];
  const float* Wt3  = (const float*)d_in[6];
  const float* bt3  = (const float*)d_in[7];
  const float* gam  = (const float*)d_in[8];
  const float* bet  = (const float*)d_in[9];
  float* out = (float*)d_out;

  char* ws = (char*)d_ws;
  size_t off = 0;
  auto take = [&](size_t bytes) -> char* {
    char* p = ws + off;
    off = (off + bytes + 255) & ~(size_t)255;
    return p;
  };
  _Float16* Wt1hT = (_Float16*)take((size_t)KW * 2 * CT * 2);
  _Float16* Wt3hT = (_Float16*)take((size_t)KW * 2 * COO * 2);
  _Float16* WshT  = (_Float16*)take((size_t)KW * CT * 2);
  _Float16* gkT   = (_Float16*)take((size_t)KSS * NN * NN * 2);
  _Float16* x1T   = (_Float16*)take((size_t)BB * T1 * CT * NN * 2);
  _Float16* x2h   = (_Float16*)take((size_t)BB * T1 * NN * CT * 2);
  float*    x3    = (float*)take((size_t)BB * T2 * NN * COO * 4);
  float*    st    = (float*)take((size_t)BB * T2 * 2 * 4);

  k_prep_weights<<<(KW * 2 * COO + 255) / 256, 256, 0, stream>>>(Wt1, Wt3, Ws, Wt1hT, Wt3hT, WshT);
  k_transpose_gk<<<(NN * 768) / 256, 256, 0, stream>>>(gk, gkT);                 // 3072 blocks
  k_tglu1<<<(BB * T1 * 64) / 8, 256, 0, stream>>>(x, Wt1hT, bt1, x1T);           // 3840 blocks
  k_graph<<<BB * T1 * 8, 256, 0, stream>>>(gkT, x1T, WshT, bs, x2h);             // 3840 blocks
  k_tglu2<<<(BB * T2 * 64 * 2) / 8, 256, 0, stream>>>(x2h, Wt3hT, bt3, x3);      // 7168 blocks
  k_ln_stats<<<BB * T2, 256, 0, stream>>>(x3, st);
  size_t total = (size_t)BB * T2 * NN * COO;
  k_ln_apply<<<(unsigned)(total / 4 / 256), 256, 0, stream>>>(x3, st, gam, bet, out);
}
